// CorrFusion_34016140984914
// MI455X (gfx1250) — compile-verified, matrix operand order checked by reading, not run on hardware
//
#include <hip/hip_runtime.h>
#include <hip/hip_bf16.h>
#include <cstdint>
#include <cstddef>

// ---------------------------------------------------------------------------
// Types for CDNA5 WMMA (wave32)
// ---------------------------------------------------------------------------
typedef __bf16 bf16x16 __attribute__((ext_vector_type(16)));
typedef float  f32x8   __attribute__((ext_vector_type(8)));
typedef int    v4i_t   __attribute__((ext_vector_type(4)));

#define BM 128
#define BN 128
#define BK 32
#define LDA 40   // padded LDS row stride (shorts): 80B rows keep 16B alignment

#define GLOBAL_AS __attribute__((address_space(1)))
#define LDS_AS    __attribute__((address_space(3)))

#if __has_builtin(__builtin_amdgcn_global_load_async_to_lds_b128) && \
    __has_builtin(__builtin_amdgcn_s_wait_asynccnt)
#define HAVE_ASYNC_LDS 1
#else
#define HAVE_ASYNC_LDS 0
#endif

__device__ __forceinline__ unsigned short f32_to_bf16(float f) {
  unsigned int u = __float_as_uint(f);
  u += 0x7FFFu + ((u >> 16) & 1u);   // round-to-nearest-even
  return (unsigned short)(u >> 16);
}

// Load a 16x32 bf16 fragment (A-pattern) from LDS.
// lane 0-15: row = lane, K chunks [0..7],[16..23]; lane 16-31: K [8..15],[24..31]
__device__ __forceinline__ bf16x16 frag_ld(const unsigned short* s, int rowBase, int lane) {
  int r  = rowBase + (lane & 15);
  int kh = (lane >> 4) & 1;
  const unsigned short* p = s + r * LDA + kh * 8;
  union { uint4 q[2]; bf16x16 v; } u;
  u.q[0] = *(const uint4*)(p);        // K = kh*8 .. kh*8+7
  u.q[1] = *(const uint4*)(p + 16);   // K = 16+kh*8 .. 16+kh*8+7
  return u.v;
}

// ---------------------------------------------------------------------------
// GEMM:  C[M,N] = A[M,Kp] * Bt[N,Kp]^T   (bf16 in, f32 accumulate)
// Double-buffered LDS; async global->LDS staging when available (ASYNCcnt).
// Epilogue scatters to NCHW: out[(b*chanStride + n)*HW + hw] = acc + bias[n]
// ---------------------------------------------------------------------------
__global__ __launch_bounds__(256)
void gemm_bf16_wmma(const unsigned short* __restrict__ A,
                    const unsigned short* __restrict__ Bt,
                    float* __restrict__ out,
                    const float* __restrict__ bias,
                    int M, int Kp, int Nreal, int chanStride, int HW)
{
  __shared__ unsigned short As[2][BM * LDA];
  __shared__ unsigned short Bs[2][BN * LDA];

  const int tid  = threadIdx.x;
  const int lane = tid & 31;
  const int wave = tid >> 5;
  const int wm   = wave >> 2;          // 0..1  -> 64 rows of M
  const int wn   = wave & 3;           // 0..3  -> 32 cols of N
  const int mBase = blockIdx.y * BM;
  const int nBase = blockIdx.x * BN;

  f32x8 acc[4][2];
  #pragma unroll
  for (int i = 0; i < 4; ++i)
    #pragma unroll
    for (int j = 0; j < 2; ++j)
      acc[i][j] = (f32x8){0.f,0.f,0.f,0.f,0.f,0.f,0.f,0.f};

  const int nk = Kp / BK;

#if HAVE_ASYNC_LDS
  // ---- async double-buffered pipeline: stage tile kt+1 while computing kt ----
  auto stage = [&](int kt, int buf) {
    int k0 = kt * BK;
    #pragma unroll
    for (int i = 0; i < 2; ++i) {
      int idx = tid + i * 256;         // 0..511
      int row = idx >> 2;
      int qc  = idx & 3;
      __builtin_amdgcn_global_load_async_to_lds_b128(
          (GLOBAL_AS v4i_t*)(A + (size_t)(mBase + row) * Kp + k0 + qc * 8),
          (LDS_AS v4i_t*)(&As[buf][row * LDA + qc * 8]), 0, 0);
      __builtin_amdgcn_global_load_async_to_lds_b128(
          (GLOBAL_AS v4i_t*)(Bt + (size_t)(nBase + row) * Kp + k0 + qc * 8),
          (LDS_AS v4i_t*)(&Bs[buf][row * LDA + qc * 8]), 0, 0);
    }
  };
  stage(0, 0);
  for (int kt = 0; kt < nk; ++kt) {
    int buf = kt & 1;
    if (kt + 1 < nk) {
      stage(kt + 1, buf ^ 1);
      __builtin_amdgcn_s_wait_asynccnt(4);  // in-order: tile kt's 4 ops done
    } else {
      __builtin_amdgcn_s_wait_asynccnt(0);
    }
    __syncthreads();

    bf16x16 af[4], bfrag[2];
    #pragma unroll
    for (int f = 0; f < 4; ++f) af[f]    = frag_ld(As[buf], wm * 64 + f * 16, lane);
    #pragma unroll
    for (int f = 0; f < 2; ++f) bfrag[f] = frag_ld(Bs[buf], wn * 32 + f * 16, lane);
    #pragma unroll
    for (int fm = 0; fm < 4; ++fm)
      #pragma unroll
      for (int fn = 0; fn < 2; ++fn)
        acc[fm][fn] = __builtin_amdgcn_wmma_f32_16x16x32_bf16(
            false, af[fm], false, bfrag[fn], (short)0, acc[fm][fn], false, false);
    __syncthreads();
  }
#else
  // ---- fallback: synchronous staging through VGPRs ----
  for (int kt = 0; kt < nk; ++kt) {
    int k0 = kt * BK;
    #pragma unroll
    for (int i = 0; i < 2; ++i) {
      int idx = tid + i * 256;
      int row = idx >> 2;
      int qc  = idx & 3;
      uint4 va = *(const uint4*)(A  + (size_t)(mBase + row) * Kp + k0 + qc * 8);
      *(uint4*)(&As[0][row * LDA + qc * 8]) = va;
      uint4 vb = *(const uint4*)(Bt + (size_t)(nBase + row) * Kp + k0 + qc * 8);
      *(uint4*)(&Bs[0][row * LDA + qc * 8]) = vb;
    }
    if (k0 + BK < Kp) {
      __builtin_prefetch(A  + (size_t)(mBase + (tid >> 1)) * Kp + k0 + BK + (tid & 1) * 16, 0, 1);
      __builtin_prefetch(Bt + (size_t)(nBase + (tid >> 1)) * Kp + k0 + BK + (tid & 1) * 16, 0, 1);
    }
    __syncthreads();
    bf16x16 af[4], bfrag[2];
    #pragma unroll
    for (int f = 0; f < 4; ++f) af[f]    = frag_ld(As[0], wm * 64 + f * 16, lane);
    #pragma unroll
    for (int f = 0; f < 2; ++f) bfrag[f] = frag_ld(Bs[0], wn * 32 + f * 16, lane);
    #pragma unroll
    for (int fm = 0; fm < 4; ++fm)
      #pragma unroll
      for (int fn = 0; fn < 2; ++fn)
        acc[fm][fn] = __builtin_amdgcn_wmma_f32_16x16x32_bf16(
            false, af[fm], false, bfrag[fn], (short)0, acc[fm][fn], false, false);
    __syncthreads();
  }
#endif

  // Epilogue: C/D layout -> lane&15 = N, (lane>>4)*8+e = M, scatter to NCHW
  const int colHalf = lane & 15;
  const int rowSel  = lane >> 4;
  for (int fm = 0; fm < 4; ++fm)
    for (int fn = 0; fn < 2; ++fn) {
      int col = nBase + wn * 32 + fn * 16 + colHalf;
      if (col >= Nreal) continue;
      float bv = bias[col];
      #pragma unroll
      for (int e = 0; e < 8; ++e) {
        int row = mBase + wm * 64 + fm * 16 + rowSel * 8 + e;
        if (row >= M) continue;
        int bidx = row / HW;
        int hw   = row - bidx * HW;
        out[((size_t)bidx * chanStride + col) * HW + hw] = acc[fm][fn][e] + bv;
      }
    }
}

// ---------------------------------------------------------------------------
// BN(eval) + ReLU, NCHW f32
// ---------------------------------------------------------------------------
__global__ void bn_relu_k(const float* __restrict__ x, const float* __restrict__ g,
                          const float* __restrict__ be, const float* __restrict__ mu,
                          const float* __restrict__ var, float* __restrict__ y,
                          int total, int Cc, int HW)
{
  int idx = blockIdx.x * blockDim.x + threadIdx.x;
  if (idx >= total) return;
  int c = (idx / HW) % Cc;
  float s = g[c] * __frsqrt_rn(var[c] + 1e-5f);
  float v = (x[idx] - mu[c]) * s + be[c];
  y[idx] = v > 0.f ? v : 0.f;
}

// ---------------------------------------------------------------------------
// Direct conv for offset prediction (small N: 2 or 18 channels)
// ---------------------------------------------------------------------------
__global__ void offset_conv_k(const float* __restrict__ xn, const float* __restrict__ w,
                              const float* __restrict__ bias, float* __restrict__ off,
                              int Bn, int Cin, int H, int W, int Kk, int Oc)
{
  int idx = blockIdx.x * blockDim.x + threadIdx.x;
  int total = Bn * Oc * H * W;
  if (idx >= total) return;
  int wx = idx % W; int t = idx / W;
  int hy = t % H; t /= H;
  int o = t % Oc; int b = t / Oc;
  int pad = Kk / 2;
  float acc = bias[o];
  for (int c = 0; c < Cin; ++c) {
    const float* ch = xn + ((size_t)(b * Cin + c)) * H * W;
    const float* wc = w + ((size_t)(o * Cin + c)) * Kk * Kk;
    for (int ky = 0; ky < Kk; ++ky) {
      int y = hy - pad + ky;
      if (y < 0 || y >= H) continue;
      for (int kx = 0; kx < Kk; ++kx) {
        int xx = wx - pad + kx;
        if (xx < 0 || xx >= W) continue;
        acc += ch[y * W + xx] * wc[ky * Kk + kx];
      }
    }
  }
  off[idx] = acc;
}

// ---------------------------------------------------------------------------
// Deformable bilinear sampling -> bf16 im2col matrix A[m, c*KK + tap]
// ---------------------------------------------------------------------------
__global__ void deform_sample_k(const float* __restrict__ xn, const float* __restrict__ off,
                                unsigned short* __restrict__ A,
                                int Bn, int Cin, int H, int W, int Kk, int Kp)
{
  int KK = Kk * Kk;
  int idx = blockIdx.x * blockDim.x + threadIdx.x;
  int total = Bn * H * W * KK;
  if (idx >= total) return;
  int tap = idx % KK; int m = idx / KK;
  int wx = m % W; int hy = (m / W) % H; int b = m / (H * W);
  int pad = Kk / 2;
  int HWl = H * W;
  float dy = off[((size_t)(b * 2 * KK + (tap * 2 + 0)) * H + hy) * W + wx];
  float dx = off[((size_t)(b * 2 * KK + (tap * 2 + 1)) * H + hy) * W + wx];
  float py = (float)(hy - pad + tap / Kk) + dy;
  float px = (float)(wx - pad + tap % Kk) + dx;
  float y0f = floorf(py), x0f = floorf(px);
  int y0 = (int)y0f, x0 = (int)x0f;
  int y1 = y0 + 1, x1 = x0 + 1;
  float wy = py - y0f, wxf = px - x0f;
  float m00 = (y0 >= 0 && y0 < H && x0 >= 0 && x0 < W) ? 1.f : 0.f;
  float m01 = (y0 >= 0 && y0 < H && x1 >= 0 && x1 < W) ? 1.f : 0.f;
  float m10 = (y1 >= 0 && y1 < H && x0 >= 0 && x0 < W) ? 1.f : 0.f;
  float m11 = (y1 >= 0 && y1 < H && x1 >= 0 && x1 < W) ? 1.f : 0.f;
  int y0c = min(max(y0, 0), H - 1), x0c = min(max(x0, 0), W - 1);
  int y1c = min(max(y1, 0), H - 1), x1c = min(max(x1, 0), W - 1);
  float w00 = (1.f - wy) * (1.f - wxf) * m00;
  float w01 = (1.f - wy) * wxf * m01;
  float w10 = wy * (1.f - wxf) * m10;
  float w11 = wy * wxf * m11;
  int o00 = y0c * W + x0c, o01 = y0c * W + x1c, o10 = y1c * W + x0c, o11 = y1c * W + x1c;
  const float* img = xn + (size_t)b * Cin * HWl;
  unsigned short* arow = A + (size_t)m * Kp + tap;
  for (int c = 0; c < Cin; ++c) {
    const float* ch = img + (size_t)c * HWl;
    float v = w00 * ch[o00] + w01 * ch[o01] + w10 * ch[o10] + w11 * ch[o11];
    arow[(size_t)c * KK] = f32_to_bf16(v);
  }
}

// Zero the K-padding tail of A (Kreal..Kp) so WMMA accumulates zeros there.
__global__ void zero_tail_k(unsigned short* __restrict__ A, int M, int Kp, int Kreal)
{
  int n = Kp - Kreal;
  int idx = blockIdx.x * blockDim.x + threadIdx.x;
  if (idx >= M * n) return;
  int m = idx / n, j = idx % n;
  A[(size_t)m * Kp + Kreal + j] = 0;
}

// Convert conv weights (O, Cin, k, k) f32 -> bf16 Bt[Npad, Kp] (zero padded)
__global__ void wconv_k(const float* __restrict__ w, unsigned short* __restrict__ Bt,
                        int Oc, int CinKK, int Kp, int Npad)
{
  int idx = blockIdx.x * blockDim.x + threadIdx.x;
  if (idx >= Npad * Kp) return;
  int o = idx / Kp, kp = idx % Kp;
  float v = (o < Oc && kp < CinKK) ? w[(size_t)o * CinKK + kp] : 0.f;
  Bt[idx] = f32_to_bf16(v);
}

// im2col for the 3x3 embedding conv; x is (B, C, T, H, W) -> images (b*T+t)
__global__ void emb_im2col_k(const float* __restrict__ x, unsigned short* __restrict__ A,
                             int Bn, int Cin, int T, int H, int W)
{
  const int Kc = Cin * 9;
  int idx = blockIdx.x * blockDim.x + threadIdx.x;
  int total = Bn * T * H * W * Kc;
  if (idx >= total) return;
  int kidx = idx % Kc; int m2 = idx / Kc;
  int c = kidx / 9, tap = kidx % 9;
  int ky = tap / 3, kx = tap % 3;
  int wx = m2 % W; int hy = (m2 / W) % H; int img = m2 / (H * W);
  int b = img / T, t = img % T;
  int y = hy - 1 + ky, xx = wx - 1 + kx;
  float v = 0.f;
  if (y >= 0 && y < H && xx >= 0 && xx < W)
    v = x[((((size_t)b * Cin + c) * T + t) * H + y) * W + xx];
  A[(size_t)m2 * Kc + kidx] = f32_to_bf16(v);
}

// L2-normalize across channels, NCHW, in place
__global__ void normalize_k(float* __restrict__ emb, int NImg, int E, int HW)
{
  int idx = blockIdx.x * blockDim.x + threadIdx.x;
  if (idx >= NImg * HW) return;
  int img = idx / HW, hw = idx % HW;
  float* p = emb + (size_t)img * E * HW + hw;
  float s = 0.f;
  for (int c = 0; c < E; ++c) { float v = p[(size_t)c * HW]; s += v * v; }
  float nrm = sqrtf(s);
  float inv = 1.f / (nrm > 1e-12f ? nrm : 1e-12f);
  for (int c = 0; c < E; ++c) p[(size_t)c * HW] *= inv;
}

// FlowNet-style cost volume; writes channels [256, 256+243) of feat
__global__ void corr_k(const float* __restrict__ emb, float* __restrict__ feat,
                       int Bn, int T, int E, int H, int W, int CS)
{
  int idx = blockIdx.x * blockDim.x + threadIdx.x;
  int total = Bn * (T - 1) * 81 * H * W;
  if (idx >= total) return;
  int HWl = H * W;
  int hw = idx % HWl; int t2 = idx / HWl;
  int d = t2 % 81; t2 /= 81;
  int t = t2 % (T - 1); int b = t2 / (T - 1);
  int h = hw / W, w = hw % W;
  int dy = d / 9 - 4, dx = d % 9 - 4;
  int h2 = h + dy, w2 = w + dx;
  float s = 0.f;
  if (h2 >= 0 && h2 < H && w2 >= 0 && w2 < W) {
    const float* e0 = emb + (size_t)(b * T + t) * E * HWl + hw;
    const float* e1 = emb + (size_t)(b * T + t + 1) * E * HWl + h2 * W + w2;
    for (int c = 0; c < E; ++c) s += e0[(size_t)c * HWl] * e1[(size_t)c * HWl];
  }
  feat[((size_t)b * CS + 256 + t * 81 + d) * HWl + hw] = s / (float)E;
}

// ---------------------------------------------------------------------------
// Host orchestration
// ---------------------------------------------------------------------------
static inline int cdiv(int a, int b) { return (a + b - 1) / b; }

extern "C" void kernel_launch(void* const* d_in, const int* in_sizes, int n_in,
                              void* d_out, int out_size, void* d_ws, size_t ws_size,
                              hipStream_t stream)
{
  const int B = 2, Cin = 64, T = 4, H = 64, W = 64, HW = 64 * 64;
  const int FUS = 256, EMB = 128, OUT = 256, NECK = 499;
  const int M = B * HW;              // 8192
  const int M2 = B * T * HW;         // 32768
  const int NeckK = NECK * 9;        // 4491
  const int NeckKp = 4512;           // padded to 32
  const int NeckNp = 512;

  // input pointers (setup_inputs dict order)
  const float* x     = (const float*)d_in[0];
  const float* f_g   = (const float*)d_in[1];
  const float* f_be  = (const float*)d_in[2];
  const float* f_mu  = (const float*)d_in[3];
  const float* f_va  = (const float*)d_in[4];
  const float* f_ow  = (const float*)d_in[5];
  const float* f_ob  = (const float*)d_in[6];
  const float* f_w   = (const float*)d_in[7];
  const float* f_b   = (const float*)d_in[8];
  const float* emb_w = (const float*)d_in[9];
  const float* emb_b = (const float*)d_in[10];

  // workspace carve (deterministic, 256B aligned)
  char* p = (char*)d_ws;
  auto carve = [&](size_t bytes) { void* r = (void*)p; p += (bytes + 255) & ~(size_t)255; return r; };
  float* xn            = (float*)carve((size_t)B * NECK * HW * 4);
  unsigned short* Abuf = (unsigned short*)carve((size_t)M * NeckKp * 2);   // also holds emb im2col
  unsigned short* Wt   = (unsigned short*)carve((size_t)NeckNp * NeckKp * 2);
  float* offbuf        = (float*)carve((size_t)B * 18 * HW * 4);
  float* feat0         = (float*)carve((size_t)B * NECK * HW * 4);
  float* feat1         = (float*)carve((size_t)B * NECK * HW * 4);
  float* embf          = (float*)carve((size_t)B * T * EMB * HW * 4);

  const int TB = 256;

  // ---- Fusion block: BN+ReLU -> 1x1 offset conv -> sample -> WMMA GEMM ----
  {
    int tot = B * FUS * HW;
    bn_relu_k<<<cdiv(tot, TB), TB, 0, stream>>>(x, f_g, f_be, f_mu, f_va, xn, tot, FUS, HW);
    int toff = B * 2 * HW;
    offset_conv_k<<<cdiv(toff, TB), TB, 0, stream>>>(xn, f_ow, f_ob, offbuf, B, FUS, H, W, 1, 2);
    int tsam = M * 1;
    deform_sample_k<<<cdiv(tsam, TB), TB, 0, stream>>>(xn, offbuf, Abuf, B, FUS, H, W, 1, 256);
    int twc = 256 * 256;
    wconv_k<<<cdiv(twc, TB), TB, 0, stream>>>(f_w, Wt, 256, 256, 256, 256);
    gemm_bf16_wmma<<<dim3(2, M / BM), TB, 0, stream>>>(Abuf, Wt, feat0, f_b,
                                                       M, 256, FUS, NECK, HW);
  }

  // ---- Embedding conv (3x3, on raw x transposed) + normalize ----
  {
    int Kc = Cin * 9; // 576
    int t1 = M2 * Kc;
    emb_im2col_k<<<cdiv(t1, TB), TB, 0, stream>>>(x, Abuf, B, Cin, T, H, W);
    int twc = 128 * Kc;
    wconv_k<<<cdiv(twc, TB), TB, 0, stream>>>(emb_w, Wt, EMB, Kc, Kc, 128);
    gemm_bf16_wmma<<<dim3(1, M2 / BM), TB, 0, stream>>>(Abuf, Wt, embf, emb_b,
                                                        M2, Kc, EMB, EMB, HW);
    int tn = B * T * HW;
    normalize_k<<<cdiv(tn, TB), TB, 0, stream>>>(embf, B * T, EMB, HW);
  }

  // ---- Correlation cost volume -> feat0 channels [256, 499) ----
  {
    int tc = B * (T - 1) * 81 * HW;
    corr_k<<<cdiv(tc, TB), TB, 0, stream>>>(embf, feat0, B, T, EMB, H, W, NECK);
  }

  // ---- Neck: 6 deformable conv blocks (3x3) via WMMA GEMM ----
  float* cur = feat0;
  float* nxt = feat1;
  for (int l = 0; l < 6; ++l) {
    int base = 11 + l * 8;
    const float* g  = (const float*)d_in[base + 0];
    const float* be = (const float*)d_in[base + 1];
    const float* mu = (const float*)d_in[base + 2];
    const float* va = (const float*)d_in[base + 3];
    const float* ow = (const float*)d_in[base + 4];
    const float* ob = (const float*)d_in[base + 5];
    const float* w  = (const float*)d_in[base + 6];
    const float* bb = (const float*)d_in[base + 7];
    int Oc = (l == 5) ? OUT : NECK;
    float* outp = (l == 5) ? (float*)d_out : nxt;

    int tot = B * NECK * HW;
    bn_relu_k<<<cdiv(tot, TB), TB, 0, stream>>>(cur, g, be, mu, va, xn, tot, NECK, HW);
    int toff = B * 18 * HW;
    offset_conv_k<<<cdiv(toff, TB), TB, 0, stream>>>(xn, ow, ob, offbuf, B, NECK, H, W, 3, 18);
    int tz = M * (NeckKp - NeckK);
    zero_tail_k<<<cdiv(tz, TB), TB, 0, stream>>>(Abuf, M, NeckKp, NeckK);
    int tsam = M * 9;
    deform_sample_k<<<cdiv(tsam, TB), TB, 0, stream>>>(xn, offbuf, Abuf, B, NECK, H, W, 3, NeckKp);
    int twc = NeckNp * NeckKp;
    wconv_k<<<cdiv(twc, TB), TB, 0, stream>>>(w, Wt, Oc, NeckK, NeckKp, NeckNp);
    gemm_bf16_wmma<<<dim3(NeckNp / BN, M / BM), TB, 0, stream>>>(Abuf, Wt, outp, bb,
                                                                 M, NeckKp, Oc, Oc, HW);
    float* tmp = cur; cur = nxt; nxt = tmp;
  }
}
